// STLSTMClassifier_57793079935484
// MI455X (gfx1250) — matrix operand order repaired and checked by reading, not compiled, not run on hardware
//
#include <hip/hip_runtime.h>
#include <hip/hip_bf16.h>
#include <math.h>

// Problem dims
#define BB 128
#define TT 100
#define DD 256
#define HH 512
#define LL 40000
#define NG 32            // HH/16 column-slice workgroups for the recurrence

typedef __bf16 bf16;
typedef bf16  bf8_v  __attribute__((ext_vector_type(8)));
typedef bf16  bf16_v __attribute__((ext_vector_type(16)));
typedef float v8f    __attribute__((ext_vector_type(8)));
typedef int   i32x4  __attribute__((ext_vector_type(4)));

#if __has_builtin(__builtin_amdgcn_global_load_async_to_lds_b128)
#define HAVE_ASYNC_LDS 1
#endif

// ---- WMMA fragment helpers (wave32, v_wmma_f32_16x16x32_bf16) -------------
// A (16x32, M x K): lanes 0-15 = rows, K contiguous 16 starting at (lane<16?0:16)
__device__ __forceinline__ bf16_v load_a_frag(const bf16* p) {
  return *(const bf16_v*)(p);
}
// B (32x16, K x N): lane n=lane&15, K contiguous 16 starting at (lane<16?0:16)
__device__ __forceinline__ bf16_v load_b_frag(const bf16* p) {
  return *(const bf16_v*)(p);
}
__device__ __forceinline__ v8f wmma_bf16(bf16_v a, bf16_v b, v8f c) {
  return __builtin_amdgcn_wmma_f32_16x16x32_bf16(false, a, false, b, (short)0, c, false, false);
}
__device__ __forceinline__ float sigmoidf_(float x) { return 1.f / (1.f + __expf(-x)); }

// 16-byte global->LDS stage: async (ASYNCcnt, no VGPR data) when available
__device__ __forceinline__ void stage16(const bf16* g, bf16* l) {
#ifdef HAVE_ASYNC_LDS
  __builtin_amdgcn_global_load_async_to_lds_b128(
      (__attribute__((address_space(1))) i32x4*)g,
      (__attribute__((address_space(3))) i32x4*)l, 0, 0);
#else
  *(bf8_v*)l = *(const bf8_v*)g;
#endif
}
__device__ __forceinline__ void stage_wait() {
#ifdef HAVE_ASYNC_LDS
#if __has_builtin(__builtin_amdgcn_s_wait_asynccnt)
  __builtin_amdgcn_s_wait_asynccnt(0);
#else
  asm volatile("s_wait_asynccnt 0" ::: "memory");
#endif
#endif
}

// ---- Prep: embedding gather + slot interpolation into bf16, time-major ----
__global__ void k_prep_embed(const int* __restrict__ batch_l,
                             const float* __restrict__ t_ld, const float* __restrict__ t_hd,
                             const int* __restrict__ t_l,  const int* __restrict__ t_h,
                             const float* __restrict__ d_ld, const float* __restrict__ d_hd,
                             const int* __restrict__ d_l,  const int* __restrict__ d_h,
                             const float* __restrict__ embed_l, const float* __restrict__ embed_s,
                             const float* __restrict__ embed_q,
                             bf16* __restrict__ Xl, bf16* __restrict__ Xsq)
{
  int bt = blockIdx.x;              // b*T + t  (input layout is (B,T))
  int b = bt / TT, t = bt % TT;
  int d = threadIdx.x;              // 0..255
  int li = batch_l[bt];
  Xl[((size_t)t * BB + b) * DD + d] = (bf16)embed_l[(size_t)li * DD + d];
  float xs = embed_s[(size_t)d_l[bt] * DD + d] * d_hd[bt]
           + embed_s[(size_t)d_h[bt] * DD + d] * d_ld[bt];
  float xq = embed_q[(size_t)t_l[bt] * DD + d] * t_hd[bt]
           + embed_q[(size_t)t_h[bt] * DD + d] * t_ld[bt];
  size_t base = ((size_t)t * BB + b) * (2 * DD);
  Xsq[base + d]      = (bf16)xs;
  Xsq[base + DD + d] = (bf16)xq;
}

// ---- Prep: permuted bf16 gate weights.  Wg[(wg*4+gate)*16+jj][0..767] =
//      [w_ih(row,0..255) | w_hh(row,0..511)], row = gate*H + wg*16 + jj -----
__global__ void k_prep_wg(const float* __restrict__ w_ih, const float* __restrict__ w_hh,
                          bf16* __restrict__ Wg)
{
  for (size_t i = (size_t)blockIdx.x * blockDim.x + threadIdx.x;
       i < (size_t)4 * HH * (DD + HH); i += (size_t)gridDim.x * blockDim.x) {
    int k  = (int)(i % (DD + HH));
    int rp = (int)(i / (DD + HH));
    int wg = rp >> 6, within = rp & 63;
    int gate = within >> 4, jj = within & 15;
    int col = gate * HH + wg * 16 + jj;
    float v = (k < DD) ? w_ih[(size_t)col * DD + k] : w_hh[(size_t)col * HH + (k - DD)];
    Wg[i] = (bf16)v;
  }
}

// ---- Prep: permuted bf16 slot weights. Wsq[(wg*3+s)*16+jj][0..511] =
//      [w_s(row,0..255) | w_q(row,0..255)], row = s*H + wg*16 + jj ----------
__global__ void k_prep_wsq(const float* __restrict__ w_s, const float* __restrict__ w_q,
                           bf16* __restrict__ Wsq)
{
  for (size_t i = (size_t)blockIdx.x * blockDim.x + threadIdx.x;
       i < (size_t)3 * HH * (2 * DD); i += (size_t)gridDim.x * blockDim.x) {
    int k  = (int)(i % (2 * DD));
    int rp = (int)(i / (2 * DD));
    int wg = rp / 48, within = rp % 48;
    int s = within >> 4, jj = within & 15;
    int col = s * HH + wg * 16 + jj;
    float v = (k < DD) ? w_s[(size_t)col * DD + k] : w_q[(size_t)col * DD + (k - DD)];
    Wsq[i] = (bf16)v;
  }
}

__global__ void k_prep_wout(const float* __restrict__ w_out, bf16* __restrict__ Wo)
{
  for (size_t i = (size_t)blockIdx.x * blockDim.x + threadIdx.x;
       i < (size_t)LL * HH; i += (size_t)gridDim.x * blockDim.x)
    Wo[i] = (bf16)w_out[i];
}

__global__ void k_prep_misc(const float* __restrict__ b_ih, const float* __restrict__ b_hh,
                            float* __restrict__ bsum, bf16* __restrict__ Hbuf,
                            int* __restrict__ cnt)
{
  int idx = blockIdx.x * blockDim.x + threadIdx.x;  // 131072 threads = 2*B*H
  Hbuf[idx] = (bf16)0.f;
  if (idx < 4 * HH) bsum[idx] = b_ih[idx] + b_hh[idx];
  if (idx == 0) *cnt = 0;
}

// ---- Persistent recurrence: 32 WGs x 256 threads, c-state in registers ----
__global__ void __launch_bounds__(256) k_lstm(
    const bf16* __restrict__ Xl, const bf16* __restrict__ Xsq,
    const bf16* __restrict__ Wg, const bf16* __restrict__ Wsq,
    const float* __restrict__ bsum, bf16* __restrict__ Hbuf, int* __restrict__ cnt)
{
  __shared__ __align__(32) bf16 ldsWg[64 * (DD + HH)];   // 96 KB
  __shared__ __align__(32) bf16 ldsWsq[48 * (2 * DD)];   // 48 KB
  const int wgid = blockIdx.x;

  {  // stage this WG's weight slices into LDS (async global->LDS, 16B units)
    const bf16* s0 = Wg + (size_t)wgid * 64 * (DD + HH);
    for (int i = threadIdx.x; i < 64 * (DD + HH) / 8; i += 256)
      stage16(s0 + (size_t)i * 8, ldsWg + (size_t)i * 8);
    const bf16* s1 = Wsq + (size_t)wgid * 48 * (2 * DD);
    for (int i = threadIdx.x; i < 48 * (2 * DD) / 8; i += 256)
      stage16(s1 + (size_t)i * 8, ldsWsq + (size_t)i * 8);
    stage_wait();
  }
  __syncthreads();

  const int wave = threadIdx.x >> 5;
  const int lane = threadIdx.x & 31;
  const int lo16 = lane & 15;
  const int hi   = lane >> 4;
  const int m0   = wave * 16;             // batch M-tile for this wave
  const int arow = m0 + lo16;             // batch row this lane loads for A
  const int col  = wgid * 16 + lo16;      // h-column this lane outputs

  const float bi  = bsum[0 * HH + col];
  const float bff = bsum[1 * HH + col];
  const float bg  = bsum[2 * HH + col];
  const float bo  = bsum[3 * HH + col];

  float creg[8];
#pragma unroll
  for (int r = 0; r < 8; ++r) creg[r] = 0.f;

#pragma unroll 1
  for (int t = 0; t < TT; ++t) {
    const bf16* hsrc = Hbuf + (size_t)(t & 1) * BB * HH;
    bf16*       hdst = Hbuf + (size_t)((t + 1) & 1) * BB * HH;

    v8f ai  = {0,0,0,0,0,0,0,0}, af  = {0,0,0,0,0,0,0,0};
    v8f ag  = {0,0,0,0,0,0,0,0}, ao  = {0,0,0,0,0,0,0,0};
    v8f asi = {0,0,0,0,0,0,0,0}, asf = {0,0,0,0,0,0,0,0}, aso = {0,0,0,0,0,0,0,0};

    const bf16* xlrow = Xl  + ((size_t)t * BB + arow) * DD;
    const bf16* hrow  = hsrc + (size_t)arow * HH;
    const bf16* sqrow = Xsq + ((size_t)t * BB + arow) * (2 * DD);
    __builtin_prefetch(Xl + ((size_t)(t + 1) * BB + arow) * DD, 0, 1);

#pragma unroll
    for (int kc = 0; kc < DD / 32; ++kc) {          // input projection, K=0..255
      bf16_v a = load_a_frag(xlrow + kc * 32 + hi * 16);
      const bf16* wb = ldsWg + (size_t)lo16 * (DD + HH) + kc * 32 + hi * 16;
      ai = wmma_bf16(a, load_b_frag(wb + (size_t)0 * 16 * (DD + HH)), ai);
      af = wmma_bf16(a, load_b_frag(wb + (size_t)1 * 16 * (DD + HH)), af);
      ag = wmma_bf16(a, load_b_frag(wb + (size_t)2 * 16 * (DD + HH)), ag);
      ao = wmma_bf16(a, load_b_frag(wb + (size_t)3 * 16 * (DD + HH)), ao);
    }
#pragma unroll 4
    for (int kc = 0; kc < HH / 32; ++kc) {          // recurrent projection, K=256..767
      bf16_v a = load_a_frag(hrow + kc * 32 + hi * 16);
      const bf16* wb = ldsWg + (size_t)lo16 * (DD + HH) + DD + kc * 32 + hi * 16;
      ai = wmma_bf16(a, load_b_frag(wb + (size_t)0 * 16 * (DD + HH)), ai);
      af = wmma_bf16(a, load_b_frag(wb + (size_t)1 * 16 * (DD + HH)), af);
      ag = wmma_bf16(a, load_b_frag(wb + (size_t)2 * 16 * (DD + HH)), ag);
      ao = wmma_bf16(a, load_b_frag(wb + (size_t)3 * 16 * (DD + HH)), ao);
    }
#pragma unroll 4
    for (int kc = 0; kc < (2 * DD) / 32; ++kc) {    // slot projection (xs|xq)
      bf16_v a = load_a_frag(sqrow + kc * 32 + hi * 16);
      const bf16* wb = ldsWsq + (size_t)lo16 * (2 * DD) + kc * 32 + hi * 16;
      asi = wmma_bf16(a, load_b_frag(wb + (size_t)0 * 16 * (2 * DD)), asi);
      asf = wmma_bf16(a, load_b_frag(wb + (size_t)1 * 16 * (2 * DD)), asf);
      aso = wmma_bf16(a, load_b_frag(wb + (size_t)2 * 16 * (2 * DD)), aso);
    }

    // fused LSTM epilogue (C/D layout: VGPR r -> M = r + 8*hi, N = lane&15)
#pragma unroll
    for (int r = 0; r < 8; ++r) {
      float iv = sigmoidf_(ai[r] + bi  + asi[r]);
      float fv = sigmoidf_(af[r] + bff + asf[r]);
      float ov = sigmoidf_(ao[r] + bo  + aso[r]);
      float gt = tanhf(ag[r] + bg);
      creg[r] = fv * creg[r] + iv * gt;
      float h2 = ov * tanhf(gt);                    // faithful: tanh(cell_gate)
      hdst[(size_t)(m0 + r + hi * 8) * HH + col] = (bf16)h2;
    }

    // device-scope grid barrier between steps (32 resident blocks)
    __threadfence();
    __syncthreads();
    if (threadIdx.x == 0) {
      __hip_atomic_fetch_add(cnt, 1, __ATOMIC_ACQ_REL, __HIP_MEMORY_SCOPE_AGENT);
      const int target = NG * (t + 1);
      while (__hip_atomic_load(cnt, __ATOMIC_ACQUIRE, __HIP_MEMORY_SCOPE_AGENT) < target)
        __builtin_amdgcn_s_sleep(1);
    }
    __syncthreads();
  }
}

// ---- Classifier: out = hT @ w_out^T + b_out; w_out streamed exactly once --
__global__ void __launch_bounds__(256) k_cls(const bf16* __restrict__ Hfin,
                                             const bf16* __restrict__ Wo,
                                             const float* __restrict__ b_out,
                                             float* __restrict__ out)
{
  const int wave = threadIdx.x >> 5;
  const int lane = threadIdx.x & 31;
  const int lo16 = lane & 15, hi = lane >> 4;
  const int nt = blockIdx.x * 8 + wave;     // N-tile id (16 cols each)
  if (nt * 16 >= LL) return;                // wave-uniform guard
  const int col = nt * 16 + lo16;

  v8f acc[8];
#pragma unroll
  for (int m = 0; m < 8; ++m) acc[m] = (v8f){0,0,0,0,0,0,0,0};

  const bf16* wrow = Wo + (size_t)col * HH;
#pragma unroll 4
  for (int kc = 0; kc < HH / 32; ++kc) {
    bf16_v b = load_b_frag(wrow + kc * 32 + hi * 16);   // held across all M-tiles
#pragma unroll
    for (int mt = 0; mt < 8; ++mt) {
      bf16_v a = load_a_frag(Hfin + (size_t)(mt * 16 + lo16) * HH + kc * 32 + hi * 16);
      acc[mt] = wmma_bf16(a, b, acc[mt]);
    }
  }
  const float bo = b_out[col];
#pragma unroll
  for (int mt = 0; mt < 8; ++mt)
#pragma unroll
    for (int r = 0; r < 8; ++r)
      out[(size_t)(mt * 16 + r + hi * 8) * LL + col] = acc[mt][r] + bo;
}

// ---------------------------------------------------------------------------
extern "C" void kernel_launch(void* const* d_in, const int* in_sizes, int n_in,
                              void* d_out, int out_size, void* d_ws, size_t ws_size,
                              hipStream_t stream)
{
  (void)in_sizes; (void)n_in; (void)out_size; (void)ws_size;
  const int*   batch_l = (const int*)  d_in[0];
  const float* t_ld    = (const float*)d_in[1];
  const float* t_hd    = (const float*)d_in[2];
  const int*   t_l     = (const int*)  d_in[3];
  const int*   t_h     = (const int*)  d_in[4];
  const float* d_ld    = (const float*)d_in[5];
  const float* d_hd    = (const float*)d_in[6];
  const int*   d_l     = (const int*)  d_in[7];
  const int*   d_h     = (const int*)  d_in[8];
  const float* embed_l = (const float*)d_in[9];
  const float* embed_s = (const float*)d_in[10];
  const float* embed_q = (const float*)d_in[11];
  const float* w_ih    = (const float*)d_in[12];
  const float* w_hh    = (const float*)d_in[13];
  const float* w_s     = (const float*)d_in[14];
  const float* w_q     = (const float*)d_in[15];
  const float* b_ih    = (const float*)d_in[16];
  const float* b_hh    = (const float*)d_in[17];
  const float* w_out   = (const float*)d_in[18];
  const float* b_out   = (const float*)d_in[19];
  float* out = (float*)d_out;

  // scratch layout (all offsets 256B aligned), total ~62.6 MiB
  char* ws = (char*)d_ws;
  bf16*  Xl   = (bf16*)(ws);               // T*B*D     bf16 =  6,553,600 B
  bf16*  Xsq  = (bf16*)(ws +  6553600);    // T*B*2D    bf16 = 13,107,200 B
  bf16*  Wg   = (bf16*)(ws + 19660800);    // 4H*(D+H)  bf16 =  3,145,728 B
  bf16*  Wsq  = (bf16*)(ws + 22806528);    // 3H*2D     bf16 =  1,572,864 B
  bf16*  Wo   = (bf16*)(ws + 24379392);    // L*H       bf16 = 40,960,000 B
  float* bsum = (float*)(ws + 65339392);   // 4H f32    =       8,192 B
  bf16*  Hbuf = (bf16*)(ws + 65347584);    // 2*B*H     bf16 =    262,144 B
  int*   cnt  = (int*)  (ws + 65609728);   // grid-barrier counter

  k_prep_embed<<<BB * TT, DD, 0, stream>>>(batch_l, t_ld, t_hd, t_l, t_h,
                                           d_ld, d_hd, d_l, d_h,
                                           embed_l, embed_s, embed_q, Xl, Xsq);
  k_prep_wg  <<<1536, 256, 0, stream>>>(w_ih, w_hh, Wg);
  k_prep_wsq <<<768,  256, 0, stream>>>(w_s, w_q, Wsq);
  k_prep_wout<<<4096, 256, 0, stream>>>(w_out, Wo);
  k_prep_misc<<<512,  256, 0, stream>>>(b_ih, b_hh, bsum, Hbuf, cnt);

  k_lstm<<<NG, 256, 0, stream>>>(Xl, Xsq, Wg, Wsq, bsum, Hbuf, cnt);
  // T=100 even -> final h lands in Hbuf slot 0
  k_cls<<<(LL / 16 + 7) / 8, 256, 0, stream>>>(Hbuf, Wo, b_out, out);
}
